// DifferentiableSuperpixelTokenizer_4501125726672
// MI455X (gfx1250) — matrix operand order repaired and checked by reading, not compiled, not run on hardware
//
#include <hip/hip_runtime.h>
#include <hip/hip_bf16.h>
#include <stdint.h>

// ---------------- problem constants ----------------
constexpr int B_ = 8, H_ = 224, W_ = 224, S_ = 196, D_ = 768;
constexpr int HW_ = H_ * W_;            // 50176
constexpr int PIX_ = B_ * HW_;          // 401408
constexpr int C1_ = 64;                 // conv1 out channels
constexpr int ROWS_ = B_ * S_;          // 1568 (= 98*16, exact M tiles)
constexpr int KF_ = 2304;               // fused GEMM K = 1536 + 768
constexpr int HP_ = H_ + 4, WP_ = W_ + 4;  // padded act plane 228x228
constexpr int NT_ = D_ / 16;            // 48 N-fragments of 16
constexpr int JC2_ = 18;                // conv2 K chunks (576/32)
constexpr int JCF_ = KF_ / 32;          // fused K chunks (72)
constexpr float EPS_ = 1e-5f;

typedef __attribute__((ext_vector_type(16))) __bf16 v16bf;
typedef __attribute__((ext_vector_type(16))) unsigned short v16us;
typedef __attribute__((ext_vector_type(8))) unsigned short v8us;
typedef __attribute__((ext_vector_type(8))) float v8f;

#define DEV __device__ __forceinline__

DEV unsigned short f2bf(float f) {
  unsigned u = __builtin_bit_cast(unsigned, f);
  unsigned r = (u + 0x7FFFu + ((u >> 16) & 1u)) >> 16;
  return (unsigned short)r;
}
DEV float bf2f(unsigned short h) {
  unsigned u = ((unsigned)h) << 16;
  return __builtin_bit_cast(float, u);
}
DEV v16us concat8(v8us a, v8us b) {
  return __builtin_shufflevector(a, b, 0, 1, 2, 3, 4, 5, 6, 7, 8, 9, 10, 11, 12, 13, 14, 15);
}
DEV v8f wmma_bf16(v16us a, v16us b, v8f c) {
  return __builtin_amdgcn_wmma_f32_16x16x32_bf16(
      false, __builtin_bit_cast(v16bf, a), false, __builtin_bit_cast(v16bf, b),
      (short)0, c, false, false);
}

// ---------------- conv1: 3->64, 3x3, pad 1 ----------------
__global__ void conv1_kernel(const float* __restrict__ img,
                             const float* __restrict__ w,
                             const float* __restrict__ bias,
                             float* __restrict__ out) {
  __shared__ float sw[C1_ * 27];
  __shared__ float sb[C1_];
  int t = threadIdx.x;
  for (int i = t; i < C1_ * 27; i += 256) sw[i] = w[i];
  if (t < C1_) sb[t] = bias[t];
  __syncthreads();
  int pix = blockIdx.x * 256 + t;
  if (pix >= PIX_) return;
  int b = pix / HW_;
  int r = pix % HW_;
  int y = r / W_, x = r % W_;
  float in[27];
#pragma unroll
  for (int c = 0; c < 3; c++)
#pragma unroll
    for (int ky = 0; ky < 3; ky++)
#pragma unroll
      for (int kx = 0; kx < 3; kx++) {
        int yy = y + ky - 1, xx = x + kx - 1;
        float v = 0.f;
        if (yy >= 0 && yy < H_ && xx >= 0 && xx < W_)
          v = img[((b * 3 + c) * H_ + yy) * W_ + xx];
        in[c * 9 + ky * 3 + kx] = v;
      }
  for (int o = 0; o < C1_; o++) {
    float acc = sb[o];
#pragma unroll
    for (int i = 0; i < 27; i++) acc += sw[o * 27 + i] * in[i];
    out[(((size_t)b * C1_ + o) * H_ + y) * W_ + x] = acc;
  }
}

// ---------------- BN stats over f32 NCHW planes ----------------
__global__ void stats_nchw_f32(const float* __restrict__ x, float* __restrict__ sum,
                               float* __restrict__ sumsq) {
  int bc = blockIdx.x;  // b*64 + c
  int c = bc % C1_;
  const float* p = x + (size_t)bc * HW_;
  float s = 0.f, q = 0.f;
  for (int i = threadIdx.x; i < HW_; i += 256) {
    float v = p[i];
    s += v;
    q += v * v;
  }
  __shared__ float ss[256], qq[256];
  ss[threadIdx.x] = s;
  qq[threadIdx.x] = q;
  __syncthreads();
  for (int st = 128; st > 0; st >>= 1) {
    if (threadIdx.x < st) {
      ss[threadIdx.x] += ss[threadIdx.x + st];
      qq[threadIdx.x] += qq[threadIdx.x + st];
    }
    __syncthreads();
  }
  if (threadIdx.x == 0) {
    atomicAdd(&sum[c], ss[0]);
    atomicAdd(&sumsq[c], qq[0]);
  }
}

// ---------------- BN finalize -> per-channel (scale, shift) ----------------
__global__ void bn_finalize(const float* __restrict__ sum, const float* __restrict__ sumsq,
                            const float* __restrict__ g, const float* __restrict__ bb,
                            float* __restrict__ param, int nch, float invcount) {
  int c = blockIdx.x * blockDim.x + threadIdx.x;
  if (c >= nch) return;
  float m = sum[c] * invcount;
  float v = sumsq[c] * invcount - m * m;
  float rs = rsqrtf(v + EPS_);
  float sc = g[c] * rs;
  param[2 * c] = sc;
  param[2 * c + 1] = bb[c] - m * sc;
}

// ---- NCHW f32 -> padded NHWC bf16 activations, BN1+ReLU fused (LDS transpose) ----
__global__ void act1_transpose(const float* __restrict__ cin, const float* __restrict__ bn1p,
                               unsigned short* __restrict__ act) {
  __shared__ unsigned short tile[W_ * C1_];  // 28 KB
  int b = blockIdx.x / H_;
  int y = blockIdx.x % H_;
  int t = threadIdx.x;
  for (int idx = t; idx < W_ * C1_; idx += 256) {
    int c = idx / W_, x = idx % W_;  // coalesced over x
    float v = cin[(((size_t)b * C1_ + c) * H_ + y) * W_ + x];
    v = v * bn1p[2 * c] + bn1p[2 * c + 1];
    v = v > 0.f ? v : 0.f;
    tile[x * C1_ + c] = f2bf(v);
  }
  __syncthreads();
  unsigned short* dst = act + (((size_t)b * HP_ + (y + 2)) * WP_ + 2) * C1_;
  const uint4* srcv = (const uint4*)tile;
  uint4* dstv = (uint4*)dst;
  for (int vv = t; vv < W_ * C1_ / 8; vv += 256) dstv[vv] = srcv[vv];
}

// ---- pack conv2 weights fragment-major: wfrag[nt16][j][lane][16] ----
__global__ void pack_w1_frag(const float* __restrict__ w2, unsigned short* __restrict__ wp) {
  int idx = blockIdx.x * 256 + threadIdx.x;  // 48*18*32*16 = 442368
  if (idx >= NT_ * JC2_ * 32 * 16) return;
  int e = idx & 15;
  int lane = (idx >> 4) & 31;
  int j = (idx >> 9) % JC2_;
  int nt16 = idx / (16 * 32 * JC2_);
  int k = j * 32 + ((e >> 3) << 4) + (lane >> 4) * 8 + (e & 7);  // K index, k = kyx*64 + c
  int c = k & 63, kyx = k >> 6;
  int o = nt16 * 16 + (lane & 15);
  wp[idx] = f2bf(w2[((size_t)o * C1_ + c) * 9 + kyx]);
}

// ------- conv2 implicit GEMM, 2x2 register-blocked LDS-free WMMA inner loop -------
__global__ void __launch_bounds__(128) conv2_wmma(const unsigned short* __restrict__ act,
                                                  const unsigned short* __restrict__ wfrag,
                                                  const float* __restrict__ bias2,
                                                  unsigned short* __restrict__ out) {
  __shared__ float Dt[32 * 128];  // 16 KB
  int t = threadIdx.x;
  int wave = t >> 5, lane = t & 31;
  int half = lane >> 4, ml = lane & 15;
  int pix0 = blockIdx.x * 32;
  int bb = pix0 / HW_;
  int rr = pix0 % HW_;
  int y = rr / W_, x0 = rr % W_;  // W divisible by 32 -> tile on one row
  int nt0 = blockIdx.y * 8 + wave * 2;
  const unsigned short* abase =
      act + (((size_t)bb * HP_ + y) * WP_ + (x0 + ml)) * C1_ + half * 8;
  const unsigned short* bbase = wfrag + ((size_t)nt0 * JC2_ * 32 + lane) * 16;
  constexpr int BSTRIDE = JC2_ * 32 * 16;  // elems between adjacent nt16 streams
  __builtin_prefetch(bbase, 0, 3);
  __builtin_prefetch(bbase + BSTRIDE, 0, 3);
  v8f acc00 = {}, acc01 = {}, acc10 = {}, acc11 = {};
#pragma unroll
  for (int j = 0; j < JC2_; j++) {
    int kyx = j >> 1;
    int ky = kyx / 3, kx = kyx % 3;
    int c0 = (j & 1) * 32;
    int aoff = ((2 * ky) * WP_ + 2 * kx) * C1_ + c0;  // constant after unroll
    v16us A0 = concat8(*(const v8us*)(abase + aoff),
                       *(const v8us*)(abase + aoff + 16));
    v16us A1 = concat8(*(const v8us*)(abase + aoff + 16 * C1_),
                       *(const v8us*)(abase + aoff + 16 * C1_ + 16));
    v16us B0 = concat8(*(const v8us*)(bbase + (size_t)j * 512),
                       *(const v8us*)(bbase + (size_t)j * 512 + 8));
    v16us B1 = concat8(*(const v8us*)(bbase + BSTRIDE + (size_t)j * 512),
                       *(const v8us*)(bbase + BSTRIDE + (size_t)j * 512 + 8));
    acc00 = wmma_bf16(A0, B0, acc00);
    acc01 = wmma_bf16(A0, B1, acc01);
    acc10 = wmma_bf16(A1, B0, acc10);
    acc11 = wmma_bf16(A1, B1, acc11);
  }
  float b0v = bias2[nt0 * 16 + ml];
  float b1v = bias2[(nt0 + 1) * 16 + ml];
#pragma unroll
  for (int r = 0; r < 8; r++) {
    int m = r + 8 * half;
    Dt[m * 128 + wave * 32 + ml] = acc00[r] + b0v;
    Dt[m * 128 + wave * 32 + 16 + ml] = acc01[r] + b1v;
    Dt[(16 + m) * 128 + wave * 32 + ml] = acc10[r] + b0v;
    Dt[(16 + m) * 128 + wave * 32 + 16 + ml] = acc11[r] + b1v;
  }
  __syncthreads();
#pragma unroll
  for (int e = 0; e < 32; e++) {
    int idx = t + e * 128;  // 0..4095
    int ol = idx & 127, m = idx >> 7;
    out[(size_t)(pix0 + m) * D_ + blockIdx.y * 128 + ol] = f2bf(Dt[m * 128 + ol]);
  }
}

// ---------------- BN2 stats over NHWC bf16 ----------------
__global__ void stats_nhwc_bf16(const unsigned short* __restrict__ x,
                                float* __restrict__ sum, float* __restrict__ sumsq) {
  int t = threadIdx.x;
  int p0 = blockIdx.x * 512;
  float s[3] = {0.f, 0.f, 0.f}, q[3] = {0.f, 0.f, 0.f};
  for (int p = 0; p < 512; p++) {
    const unsigned short* row = x + (size_t)(p0 + p) * D_;
#pragma unroll
    for (int e = 0; e < 3; e++) {
      float v = bf2f(row[t + e * 256]);
      s[e] += v;
      q[e] += v * v;
    }
  }
#pragma unroll
  for (int e = 0; e < 3; e++) {
    atomicAdd(&sum[t + e * 256], s[e]);
    atomicAdd(&sumsq[t + e * 256], q[e]);
  }
}

// ---------------- segment bbox / counts ----------------
__global__ void seg_bbox(const int* __restrict__ segs, int* __restrict__ cnt,
                         int* __restrict__ mnx, int* __restrict__ mxx,
                         int* __restrict__ mny, int* __restrict__ mxy) {
  int pix = blockIdx.x * 256 + threadIdx.x;
  if (pix >= PIX_) return;
  int b = pix / HW_;
  int r = pix % HW_;
  int y = r / W_, x = r % W_;
  int row = b * S_ + segs[pix];
  atomicAdd(&cnt[row], 1);
  atomicMin(&mnx[row], x);
  atomicMax(&mxx[row], x);
  atomicMin(&mny[row], y);
  atomicMax(&mxy[row], y);
}

// ---------------- segment sum/max of BN2+ReLU features ----------------
__global__ void seg_reduce(const unsigned short* __restrict__ feat,
                           const int* __restrict__ segs, const float* __restrict__ bn2p,
                           float* __restrict__ ssum, int* __restrict__ smax_bits) {
  int pix = blockIdx.x;
  int t = threadIdx.x;
  int b = pix / HW_;
  int s = segs[pix];
  size_t row = (size_t)(b * S_ + s) * D_;
  const unsigned short* f = feat + (size_t)pix * D_;
#pragma unroll
  for (int e = 0; e < 3; e++) {
    int d = t + e * 256;
    float v = bf2f(f[d]);
    v = v * bn2p[2 * d] + bn2p[2 * d + 1];
    v = v > 0.f ? v : 0.f;
    atomicAdd(&ssum[row + d], v);
    // v >= 0 -> float bits monotone as int; 0-init matches where(occ,...,0)
    atomicMax(&smax_bits[row + d], __float_as_int(v));
  }
}

// ---------------- build combined A [rows,2304] bf16 + per-element bias ----------------
__global__ void prep_rows(const float* __restrict__ ssum, const float* __restrict__ smax,
                          const int* __restrict__ cnt, const int* __restrict__ mnx,
                          const int* __restrict__ mxx, const int* __restrict__ mny,
                          const int* __restrict__ mxy, const float* __restrict__ cent,
                          const float* __restrict__ pos_w, const float* __restrict__ pos_b,
                          const float* __restrict__ fus_b, const float* __restrict__ sw1,
                          const float* __restrict__ sb1, const float* __restrict__ sb2,
                          unsigned short* __restrict__ Apack, float* __restrict__ biasb) {
  int row = blockIdx.x;
  int t = threadIdx.x;
  int c = cnt[row];
  float count = (float)c;
  float inv = c > 0 ? 1.f / count : 1.f;  // clip(counts, 1.0)
  float area = count;
  float aspect = 1.f;
  if (c > 0) {
    float bw = fmaxf((float)(mxx[row] - mnx[row] + 1), 1e-6f);
    float bh = fmaxf((float)(mxy[row] - mny[row] + 1), 1e-6f);
    aspect = bw / bh;
  }
  float cx = cent[(size_t)row * 2 + 0] * (1.f / (float)W_);
  float cy = cent[(size_t)row * 2 + 1] * (1.f / (float)H_);
#pragma unroll
  for (int e = 0; e < 3; e++) {
    int d = t + e * 256;
    float mean = ssum[(size_t)row * D_ + d] * inv;
    float mx = smax[(size_t)row * D_ + d];
    Apack[(size_t)row * KF_ + d] = f2bf(mean);
    Apack[(size_t)row * KF_ + D_ + d] = f2bf(mx);
    float hid = area * sw1[2 * d] + aspect * sw1[2 * d + 1] + sb1[d];
    hid = hid > 0.f ? hid : 0.f;
    Apack[(size_t)row * KF_ + 2 * D_ + d] = f2bf(hid);
    biasb[(size_t)row * D_ + d] =
        cx * pos_w[2 * d] + cy * pos_w[2 * d + 1] + pos_b[d] + fus_b[d] + sb2[d];
  }
}

// ---- pack [fusion_w | shape_w2] fragment-major: wfrag2[nt16][j][lane][16] ----
__global__ void pack_wf_frag(const float* __restrict__ fw, const float* __restrict__ sw2,
                             unsigned short* __restrict__ wp) {
  int idx = blockIdx.x * 256 + threadIdx.x;  // 48*72*32*16 = 1769472
  if (idx >= NT_ * JCF_ * 32 * 16) return;
  int e = idx & 15;
  int lane = (idx >> 4) & 31;
  int j = (idx >> 9) % JCF_;
  int nt16 = idx / (16 * 32 * JCF_);
  int k = j * 32 + ((e >> 3) << 4) + (lane >> 4) * 8 + (e & 7);
  int n = nt16 * 16 + (lane & 15);
  float v = (k < 1536) ? fw[(size_t)n * 1536 + k] : sw2[(size_t)n * D_ + (k - 1536)];
  wp[idx] = f2bf(v);
}

// ---------------- fused GEMM, 1x2 register-blocked LDS-free WMMA loop ----------------
__global__ void __launch_bounds__(128) fused_wmma(const unsigned short* __restrict__ Apack,
                                                  const unsigned short* __restrict__ wfrag,
                                                  const float* __restrict__ biasb,
                                                  float* __restrict__ out) {
  int t = threadIdx.x;
  int wave = t >> 5, lane = t & 31;
  int half = lane >> 4, ml = lane & 15;
  int row0 = blockIdx.x * 16;
  int nt0 = blockIdx.y * 8 + wave * 2;
  const unsigned short* abase = Apack + (size_t)(row0 + ml) * KF_ + half * 8;
  const unsigned short* bbase = wfrag + ((size_t)nt0 * JCF_ * 32 + lane) * 16;
  constexpr int BSTRIDE = JCF_ * 32 * 16;
  __builtin_prefetch(bbase, 0, 3);
  __builtin_prefetch(bbase + BSTRIDE, 0, 3);
  v8f acc0 = {}, acc1 = {};
#pragma unroll 4
  for (int j = 0; j < JCF_; j++) {
    v16us A = concat8(*(const v8us*)(abase + j * 32),
                      *(const v8us*)(abase + j * 32 + 16));
    v16us B0 = concat8(*(const v8us*)(bbase + (size_t)j * 512),
                       *(const v8us*)(bbase + (size_t)j * 512 + 8));
    v16us B1 = concat8(*(const v8us*)(bbase + BSTRIDE + (size_t)j * 512),
                       *(const v8us*)(bbase + BSTRIDE + (size_t)j * 512 + 8));
    acc0 = wmma_bf16(A, B0, acc0);
    acc1 = wmma_bf16(A, B1, acc1);
  }
#pragma unroll
  for (int r = 0; r < 8; r++) {
    int m = r + 8 * half;
    size_t off0 = (size_t)(row0 + m) * D_ + nt0 * 16 + ml;
    out[off0] = acc0[r] + biasb[off0];
    out[off0 + 16] = acc1[r] + biasb[off0 + 16];
  }
}

// ---------------- host side ----------------
extern "C" void kernel_launch(void* const* d_in, const int* in_sizes, int n_in,
                              void* d_out, int out_size, void* d_ws, size_t ws_size,
                              hipStream_t stream) {
  const float* img = (const float*)d_in[0];
  const int* segments = (const int*)d_in[1];
  const float* centroids = (const float*)d_in[2];
  const float* conv1_w = (const float*)d_in[3];
  const float* conv1_b = (const float*)d_in[4];
  const float* bn1_g = (const float*)d_in[5];
  const float* bn1_b = (const float*)d_in[6];
  const float* conv2_w = (const float*)d_in[7];
  const float* conv2_b = (const float*)d_in[8];
  const float* bn2_g = (const float*)d_in[9];
  const float* bn2_b = (const float*)d_in[10];
  const float* fusion_w = (const float*)d_in[11];
  const float* fusion_b = (const float*)d_in[12];
  const float* pos_w = (const float*)d_in[13];
  const float* pos_b = (const float*)d_in[14];
  const float* shape_w1 = (const float*)d_in[15];
  const float* shape_b1 = (const float*)d_in[16];
  const float* shape_w2 = (const float*)d_in[17];
  const float* shape_b2 = (const float*)d_in[18];
  float* outp = (float*)d_out;

  size_t off = 0;
  auto alloc = [&](size_t bytes) -> char* {
    char* p = (char*)d_ws + off;
    off += (bytes + 255) & ~(size_t)255;
    return p;
  };
  float* conv1out = (float*)alloc((size_t)B_ * C1_ * HW_ * 4);
  unsigned short* act1 = (unsigned short*)alloc((size_t)B_ * HP_ * WP_ * C1_ * 2);
  unsigned short* conv2out = (unsigned short*)alloc((size_t)PIX_ * D_ * 2);
  unsigned short* wfrag1 = (unsigned short*)alloc((size_t)NT_ * JC2_ * 32 * 16 * 2);
  unsigned short* wfrag2 = (unsigned short*)alloc((size_t)NT_ * JCF_ * 32 * 16 * 2);
  unsigned short* Apack = (unsigned short*)alloc((size_t)ROWS_ * KF_ * 2);
  float* biasbuf = (float*)alloc((size_t)ROWS_ * D_ * 4);
  float* seg_sum = (float*)alloc((size_t)ROWS_ * D_ * 4);
  float* seg_max = (float*)alloc((size_t)ROWS_ * D_ * 4);
  float* s1sum = (float*)alloc(C1_ * 4);
  float* s1sq = (float*)alloc(C1_ * 4);
  float* bn1p = (float*)alloc(2 * C1_ * 4);
  float* s2sum = (float*)alloc(D_ * 4);
  float* s2sq = (float*)alloc(D_ * 4);
  float* bn2p = (float*)alloc(2 * D_ * 4);
  int* cnt = (int*)alloc(ROWS_ * 4);
  int* mnx = (int*)alloc(ROWS_ * 4);
  int* mxx = (int*)alloc(ROWS_ * 4);
  int* mny = (int*)alloc(ROWS_ * 4);
  int* mxy = (int*)alloc(ROWS_ * 4);

  // init accumulators + padded activation halo (graph-capturable memsets)
  hipMemsetAsync(act1, 0, (size_t)B_ * HP_ * WP_ * C1_ * 2, stream);
  hipMemsetAsync(seg_sum, 0, (size_t)ROWS_ * D_ * 4, stream);
  hipMemsetAsync(seg_max, 0, (size_t)ROWS_ * D_ * 4, stream);
  hipMemsetAsync(s1sum, 0, C1_ * 4, stream);
  hipMemsetAsync(s1sq, 0, C1_ * 4, stream);
  hipMemsetAsync(s2sum, 0, D_ * 4, stream);
  hipMemsetAsync(s2sq, 0, D_ * 4, stream);
  hipMemsetAsync(cnt, 0, ROWS_ * 4, stream);
  hipMemsetAsync(mnx, 0x7F, ROWS_ * 4, stream);
  hipMemsetAsync(mny, 0x7F, ROWS_ * 4, stream);
  hipMemsetAsync(mxx, 0, ROWS_ * 4, stream);
  hipMemsetAsync(mxy, 0, ROWS_ * 4, stream);

  // stage 1: conv1 + BN1 stats + padded NHWC bf16 activations
  conv1_kernel<<<PIX_ / 256, 256, 0, stream>>>(img, conv1_w, conv1_b, conv1out);
  stats_nchw_f32<<<B_ * C1_, 256, 0, stream>>>(conv1out, s1sum, s1sq);
  bn_finalize<<<1, 64, 0, stream>>>(s1sum, s1sq, bn1_g, bn1_b, bn1p, C1_,
                                    1.f / (float)PIX_);
  act1_transpose<<<B_ * H_, 256, 0, stream>>>(conv1out, bn1p, act1);

  // stage 2: conv2 implicit GEMM (WMMA bf16, 2x2 register blocking)
  pack_w1_frag<<<(NT_ * JC2_ * 32 * 16 + 255) / 256, 256, 0, stream>>>(conv2_w, wfrag1);
  conv2_wmma<<<dim3(PIX_ / 32, D_ / 128), 128, 0, stream>>>(act1, wfrag1, conv2_b,
                                                            conv2out);

  // stage 3: BN2 stats + segment pooling
  stats_nhwc_bf16<<<PIX_ / 512, 256, 0, stream>>>(conv2out, s2sum, s2sq);
  bn_finalize<<<(D_ + 255) / 256, 256, 0, stream>>>(s2sum, s2sq, bn2_g, bn2_b, bn2p,
                                                    D_, 1.f / (float)PIX_);
  seg_bbox<<<PIX_ / 256, 256, 0, stream>>>(segments, cnt, mnx, mxx, mny, mxy);
  seg_reduce<<<PIX_, 256, 0, stream>>>(conv2out, segments, bn2p, seg_sum,
                                       (int*)seg_max);

  // stage 4: fused output GEMM (fusion + shape MLP merged, pos folded into bias)
  prep_rows<<<ROWS_, 256, 0, stream>>>(seg_sum, seg_max, cnt, mnx, mxx, mny, mxy,
                                       centroids, pos_w, pos_b, fusion_b, shape_w1,
                                       shape_b1, shape_b2, Apack, biasbuf);
  pack_wf_frag<<<(NT_ * JCF_ * 32 * 16 + 255) / 256, 256, 0, stream>>>(fusion_w,
                                                                       shape_w2, wfrag2);
  fused_wmma<<<dim3(ROWS_ / 16, D_ / 128), 128, 0, stream>>>(Apack, wfrag2, biasbuf,
                                                             outp);
  (void)in_sizes; (void)n_in; (void)out_size; (void)ws_size;
}
